// ParallelExpertWrapper_12060268167401
// MI455X (gfx1250) — matrix operand order repaired and checked by reading, not compiled, not run on hardware
//
#include <hip/hip_runtime.h>

// Problem constants (match reference)
constexpr int E_  = 16;
constexpr int B_  = 4096;
constexpr int DX_ = 128;
constexpr int DC_ = 64;
constexpr int DIN_ = DX_ + DC_; // 192
constexpr int DH_ = 1024;
constexpr int DO_ = 128;

typedef __attribute__((ext_vector_type(16))) __bf16 v16bf;
typedef __attribute__((ext_vector_type(2)))  __bf16 v2bf;
typedef __attribute__((ext_vector_type(8)))  float  v8f;

__device__ __forceinline__ unsigned pack_hi2(float a, float b) {
    v2bf p; p[0] = (__bf16)a; p[1] = (__bf16)b;
    return __builtin_bit_cast(unsigned, p);
}
__device__ __forceinline__ unsigned pack_lo2(float a, float b) {
    __bf16 ha = (__bf16)a, hb = (__bf16)b;
    v2bf p; p[0] = (__bf16)(a - (float)ha); p[1] = (__bf16)(b - (float)hb);
    return __builtin_bit_cast(unsigned, p);
}

__device__ __forceinline__ v8f wmma_bf(v16bf a, v16bf b, v8f c) {
    // v_wmma_f32_16x16x32_bf16  D = A(16x32) * B(32x16) + C
    return __builtin_amdgcn_wmma_f32_16x16x32_bf16(false, a, false, b,
                                                   (short)0, c, false, false);
}

// bf16x2 split product: A*B ~= Ah*Bh + Ah*Bl + Al*Bh (f32 accumulate)
__device__ __forceinline__ v8f mm3(v16bf ah, v16bf al, v16bf bh, v16bf bl, v8f c) {
    c = wmma_bf(ah, bh, c);
    c = wmma_bf(ah, bl, c);
    c = wmma_bf(al, bh, c);
    return c;
}

__global__ __launch_bounds__(256, 1)
void moe_mlp_fused_kernel(const float* __restrict__ x,
                          const float* __restrict__ cond,
                          const float* __restrict__ W1,
                          const float* __restrict__ b1,
                          const float* __restrict__ W2,
                          const float* __restrict__ b2,
                          float* __restrict__ out)
{
    // LDS staging (padded strides -> conflict-free fragment gathers)
    __shared__ unsigned sW1h[32 * 97];   // W1 slice [32 hcols][96 k-pairs], hi bf16
    __shared__ unsigned sW1l[32 * 97];   // lo bf16
    __shared__ unsigned sW2h[128 * 17];  // W2 slice [128 o][16 k-pairs], hi
    __shared__ unsigned sW2l[128 * 17];  // lo
    __shared__ float    sH[8][16 * 34];  // per-wave relu(h) 16x32, stride 34

    const int tid  = threadIdx.x;
    const int lane = tid & 31;
    const int wave = tid >> 5;
    const int lm   = lane & 15;      // M (or N) index within fragment
    const int hh   = lane >> 4;      // 0 = lanes 0-15, 1 = lanes 16-31
    const int e    = blockIdx.y;
    const int rowBase = blockIdx.x * 128 + wave * 16;
    const int row  = rowBase + lm;

    const float* W1e = W1 + (size_t)e * DH_ * DIN_;
    const float* W2e = W2 + (size_t)e * DO_ * DH_;
    const float* xrow = x    + ((size_t)e * B_ + row) * DX_;
    const float* crow = cond + ((size_t)e * B_ + row) * DC_;

    // ---- Preload activation A-fragments (K=192 -> 6 bf16 K-tiles), hi/lo split.
    // 16-bit A layout (ISA 7.12.2): lane holds M = lane%16;
    //   j<4 : K = 2j   + (lane>=16 ? 8 : 0)  (pair K,K+1)
    //   j>=4: K = 2j+8 + (lane>=16 ? 8 : 0)
    v16bf aXh[6], aXl[6];
    #pragma unroll
    for (int kt = 0; kt < 6; ++kt) {
        #pragma unroll
        for (int j = 0; j < 8; ++j) {
            const int k0 = kt * 32 + 2 * j + (j >= 4 ? 8 : 0) + (hh ? 8 : 0);
            float f0, f1;
            if (k0 < DX_) { f0 = xrow[k0];        f1 = xrow[k0 + 1]; }
            else          { f0 = crow[k0 - DX_];  f1 = crow[k0 - DX_ + 1]; }
            const __bf16 h0 = (__bf16)f0, h1 = (__bf16)f1;
            aXh[kt][2 * j]     = h0;
            aXh[kt][2 * j + 1] = h1;
            aXl[kt][2 * j]     = (__bf16)(f0 - (float)h0);
            aXl[kt][2 * j + 1] = (__bf16)(f1 - (float)h1);
        }
    }

    // y accumulators: 16 rows x 128 DO per wave (8 C/D tiles)
    const v8f vzero = {0.f, 0.f, 0.f, 0.f, 0.f, 0.f, 0.f, 0.f};
    v8f yacc[8];
    #pragma unroll
    for (int n = 0; n < 8; ++n) yacc[n] = vzero;

    // ---- Main loop over DH in tiles of 32 ----
    for (int ht = 0; ht < DH_ / 32; ++ht) {
        const int hbase = ht * 32;
        __syncthreads();  // protect sW1/sW2 from previous iteration readers

        // Stage W1 slice [32 x 192] as packed hi/lo bf16 pairs. 3072 pairs / 256 thr.
        for (int idx = tid; idx < 32 * 96; idx += 256) {
            const int n = idx / 96, kp = idx % 96;
            const float* p = W1e + (size_t)(hbase + n) * DIN_ + 2 * kp;
            const float f0 = p[0], f1 = p[1];
            sW1h[n * 97 + kp] = pack_hi2(f0, f1);
            sW1l[n * 97 + kp] = pack_lo2(f0, f1);
        }
        // Stage W2 slice [128 x 32]. 2048 pairs / 256 thr.
        for (int idx = tid; idx < 128 * 16; idx += 256) {
            const int o = idx / 16, kp = idx % 16;
            const float* p = W2e + (size_t)o * DH_ + hbase + 2 * kp;
            const float f0 = p[0], f1 = p[1];
            sW2h[o * 17 + kp] = pack_hi2(f0, f1);
            sW2l[o * 17 + kp] = pack_lo2(f0, f1);
        }
        // Prefetch next W1 slice rows into cache hierarchy
        if (ht + 1 < DH_ / 32 && tid < 32) {
            __builtin_prefetch(W1e + (size_t)(hbase + 32 + tid) * DIN_, 0, 1);
        }
        __syncthreads();

        // ---- GEMM1: h(16x32) = xc(16x192) * W1_slice^T, bias + relu ----
        #pragma unroll
        for (int nt = 0; nt < 2; ++nt) {
            v8f hacc = vzero;
            #pragma unroll
            for (int kt = 0; kt < 6; ++kt) {
                // B layout: lane holds N = lane%16; VGPR j -> K pair 2j (+8 for j>=4, +8 for hi half)
                v16bf bh, bl;
                #pragma unroll
                for (int j = 0; j < 8; ++j) {
                    const int kp = kt * 16 + j + (j >= 4 ? 4 : 0) + (hh ? 4 : 0);
                    const int n  = nt * 16 + lm;
                    const v2bf ph = __builtin_bit_cast(v2bf, sW1h[n * 97 + kp]);
                    const v2bf pl = __builtin_bit_cast(v2bf, sW1l[n * 97 + kp]);
                    bh[2 * j] = ph[0]; bh[2 * j + 1] = ph[1];
                    bl[2 * j] = pl[0]; bl[2 * j + 1] = pl[1];
                }
                hacc = mm3(aXh[kt], aXl[kt], bh, bl, hacc);
            }
            // bias + relu; write to per-wave LDS scratch (C/D layout -> [m][k])
            const float bias = b1[(size_t)e * DH_ + hbase + nt * 16 + lm];
            #pragma unroll
            for (int r = 0; r < 8; ++r) {
                float v = hacc[r] + bias;
                v = v > 0.f ? v : 0.f;
                sH[wave][(r + (hh ? 8 : 0)) * 34 + nt * 16 + lm] = v;
            }
        }

        // Re-gather h in A layout (same-wave LDS store->load is in-order)
        v16bf ahh, ahl;
        #pragma unroll
        for (int j = 0; j < 8; ++j) {
            const int k0 = 2 * j + (j >= 4 ? 8 : 0) + (hh ? 8 : 0);
            const float f0 = sH[wave][lm * 34 + k0];
            const float f1 = sH[wave][lm * 34 + k0 + 1];
            const __bf16 h0 = (__bf16)f0, h1 = (__bf16)f1;
            ahh[2 * j] = h0;            ahh[2 * j + 1] = h1;
            ahl[2 * j] = (__bf16)(f0 - (float)h0);
            ahl[2 * j + 1] = (__bf16)(f1 - (float)h1);
        }

        // ---- GEMM2: y(16x128) += relu(h)(16x32) * W2_slice^T ----
        #pragma unroll
        for (int nt = 0; nt < 8; ++nt) {
            v16bf bh, bl;
            #pragma unroll
            for (int j = 0; j < 8; ++j) {
                const int kp = j + (j >= 4 ? 4 : 0) + (hh ? 4 : 0);
                const int o  = nt * 16 + lm;
                const v2bf ph = __builtin_bit_cast(v2bf, sW2h[o * 17 + kp]);
                const v2bf pl = __builtin_bit_cast(v2bf, sW2l[o * 17 + kp]);
                bh[2 * j] = ph[0]; bh[2 * j + 1] = ph[1];
                bl[2 * j] = pl[0]; bl[2 * j + 1] = pl[1];
            }
            yacc[nt] = mm3(ahh, ahl, bh, bl, yacc[nt]);
        }
    }

    // ---- Epilogue: + b2, store y ----
    #pragma unroll
    for (int nt = 0; nt < 8; ++nt) {
        const int o = nt * 16 + lm;
        const float bias = b2[(size_t)e * DO_ + o];
        #pragma unroll
        for (int r = 0; r < 8; ++r) {
            const int rr = rowBase + r + (hh ? 8 : 0);
            out[((size_t)e * B_ + rr) * DO_ + o] = yacc[nt][r] + bias;
        }
    }
}

extern "C" void kernel_launch(void* const* d_in, const int* in_sizes, int n_in,
                              void* d_out, int out_size, void* d_ws, size_t ws_size,
                              hipStream_t stream) {
    const float* x    = (const float*)d_in[0];
    const float* cond = (const float*)d_in[1];
    const float* W1   = (const float*)d_in[2];
    const float* b1   = (const float*)d_in[3];
    const float* W2   = (const float*)d_in[4];
    const float* b2   = (const float*)d_in[5];
    float* out = (float*)d_out;

    dim3 grid(B_ / 128, E_);   // 32 row tiles x 16 experts
    moe_mlp_fused_kernel<<<grid, 256, 0, stream>>>(x, cond, W1, b1, W2, b2, out);
}
